// DFconvResBlock_82111184765068
// MI455X (gfx1250) — matrix operand order repaired and checked by reading, not compiled, not run on hardware
//
#include <hip/hip_runtime.h>
#include <hip/hip_bf16.h>

// ---------------------------------------------------------------------------
// DFconvResBlock for MI455X (gfx1250):
//   offset conv -> deformable conv (bf16 WMMA, f32 accum) -> 1x1 conv (WMMA)
//   -> batch BN + ReLU + residual.
// Weights pre-converted to bf16 once, then streamed into LDS per workgroup
// via the Tensor Data Mover (tensor_load_to_lds), overlapped with the
// bilinear-sampling gather phase.
// ---------------------------------------------------------------------------

typedef __attribute__((ext_vector_type(16))) __bf16 v16bf;
typedef __attribute__((ext_vector_type(8)))  float  v8f;
typedef __attribute__((ext_vector_type(4)))  unsigned u32x4;
typedef __attribute__((ext_vector_type(8)))  int      i32x8;
typedef __attribute__((ext_vector_type(4)))  int      i32x4;

#define C_CH   64
#define KKT    9
#define KTOT   576      // C_CH * KKT
#define H_IMG  192
#define W_IMG  192
#define HW     36864
#define B_N    4
#define NPIX   147456   // B_N * HW
#define TILE_PIX 32

// LDS row strides (elements). A/Y are SW-written (padded for bank spread);
// Bt/Ct are TDM-written (contiguous, no padding possible from the DMA).
#define A_STRIDE  584   // 576 + 8
#define BT_STRIDE 576
#define Y_STRIDE  72    // 64 + 8
#define CT_STRIDE 64

__device__ __forceinline__ __bf16 f2bf(float f) {
  union { float f; unsigned u; } v; v.f = f;
  unsigned r = v.u + 0x7FFFu + ((v.u >> 16) & 1u);   // round-to-nearest-even
  union { unsigned short s; __bf16 b; } o;
  o.s = (unsigned short)(r >> 16);
  return o.b;
}

#if __has_builtin(__builtin_amdgcn_tensor_load_to_lds)
#define USE_TDM 1
#else
#define USE_TDM 0
#endif

#if USE_TDM
// 1-D TDM copy: n_units8 x 8-byte units from global (gptr) into LDS (lds_off).
// D# per CDNA5 ISA ch.8: group0 = count/lds/global/type, group1 = dims.
__device__ __forceinline__ void tdm_load_1d(unsigned lds_off, const void* gptr,
                                            unsigned n_units8) {
  unsigned long long ga = (unsigned long long)gptr;
  u32x4 g0;
  g0[0] = 1u;                                   // count=1, user descriptor
  g0[1] = lds_off;                              // lds_addr (bytes)
  g0[2] = (unsigned)ga;                         // global_addr[31:0]
  g0[3] = (unsigned)(ga >> 32) | (2u << 30);    // global_addr[56:32] | type=2
  i32x8 g1;
  g1[0] = (int)(3u << 16);                      // data_size = 8B, no mask/pad
  g1[1] = (int)((n_units8 & 0xFFFFu) << 16);    // tensor_dim0[15:0]
  g1[2] = (int)(n_units8 >> 16);                // tensor_dim0[31:16], dim1=0
  g1[3] = (int)((n_units8 & 0xFFFFu) << 16);    // tile_dim0 (<=65535)
  g1[4] = 0;                                    // tile_dim1=0, tile_dim2=0
  g1[5] = (int)n_units8;                        // tensor_dim0_stride
  g1[6] = 0;
  g1[7] = 0;
  i32x4 z4 = {0, 0, 0, 0};
#if __clang_major__ >= 23
  i32x8 z8 = {0, 0, 0, 0, 0, 0, 0, 0};
  __builtin_amdgcn_tensor_load_to_lds(g0, g1, z4, z4, z8, 0);
#else
  __builtin_amdgcn_tensor_load_to_lds(g0, g1, z4, z4, 0);
#endif
}
#endif

// ---------------------------------------------------------------------------
// Kernel 0: zero the BN-statistics accumulators (128 floats).
// ---------------------------------------------------------------------------
__global__ void zero_stats_kernel(float* __restrict__ s) {
  s[threadIdx.x] = 0.0f;
}

// ---------------------------------------------------------------------------
// Kernel 1: one-shot weight conversion to bf16 in GEMM layout.
//   Wbt[o*576 + k] = bf16(W_def[o][c][kk]),  k = kk*64 + c
//   Wct[o*64  + c] = bf16(W_cat[o][c])
// grid = (64*576 + 64*64)/256 = 160 blocks.
// ---------------------------------------------------------------------------
__global__ __launch_bounds__(256) void prep_weights_kernel(
    const float* __restrict__ W_def, const float* __restrict__ W_cat,
    __bf16* __restrict__ Wbt, __bf16* __restrict__ Wct) {
  const int i = blockIdx.x * 256 + threadIdx.x;
  if (i < C_CH * KTOT) {
    const int o = i / KTOT;
    const int k = i - o * KTOT;
    const int c  = k & 63;
    const int kk = k >> 6;
    Wbt[i] = f2bf(W_def[((size_t)o * C_CH + c) * KKT + kk]);
  } else {
    const int j = i - C_CH * KTOT;               // 0..4095
    Wct[j] = f2bf(W_cat[j]);
  }
}

// ---------------------------------------------------------------------------
// Kernel 2: 3x3 offset-generating conv, 64 -> 18 channels, padding 1.
// grid = B_N * 18 * (HW/256); one block: one (b, oc) over 256 pixels.
// ---------------------------------------------------------------------------
__global__ __launch_bounds__(256) void offset_conv_kernel(
    const float* __restrict__ x, const float* __restrict__ W_off,
    const float* __restrict__ b_off, float* __restrict__ offs) {
  __shared__ float wl[KTOT];
  const int blk   = blockIdx.x;
  const int chunk = blk % (HW / 256);
  const int oc    = (blk / (HW / 256)) % 18;
  const int b     = blk / ((HW / 256) * 18);
  const int tid   = threadIdx.x;

  for (int i = tid; i < KTOT; i += 256) wl[i] = W_off[oc * KTOT + i];
  __syncthreads();

  const int hw = chunk * 256 + tid;
  const int h  = hw / W_IMG;
  const int w  = hw % W_IMG;
  float acc = b_off[oc];
  const float* xb = x + (size_t)b * C_CH * HW;
  for (int c = 0; c < C_CH; ++c) {
    const float* xc = xb + c * HW;
    const float* wc = wl + c * 9;
#pragma unroll
    for (int ki = 0; ki < 3; ++ki) {
      const int yy = h - 1 + ki;
      if (yy < 0 || yy >= H_IMG) continue;
#pragma unroll
      for (int kj = 0; kj < 3; ++kj) {
        const int xx = w - 1 + kj;
        if (xx < 0 || xx >= W_IMG) continue;
        acc += xc[yy * W_IMG + xx] * wc[ki * 3 + kj];
      }
    }
  }
  offs[((size_t)b * 18 + oc) * HW + hw] = acc;
}

// ---------------------------------------------------------------------------
// Kernel 3: TDM weight streaming + deformable sampling + two WMMA GEMMs
// + BN partial statistics. One workgroup (8 wave32) = 32 px x 64 out-ch.
// ---------------------------------------------------------------------------
__global__ __launch_bounds__(256) void deform_gemm_kernel(
    const float* __restrict__ x,     const float* __restrict__ offs,
    const __bf16* __restrict__ Wbt,  const float* __restrict__ b_def,
    const __bf16* __restrict__ Wct,  const float* __restrict__ b_cat,
    float* __restrict__ y_ws,        float* __restrict__ stat_ws) {
  extern __shared__ char smem[];
  __bf16* A_lds  = (__bf16*)smem;                       // 32 x A_STRIDE
  __bf16* Bt_lds = A_lds  + TILE_PIX * A_STRIDE;        // 64 x BT_STRIDE
  __bf16* Y_lds  = Bt_lds + C_CH * BT_STRIDE;           // 32 x Y_STRIDE
  __bf16* Ct_lds = Y_lds  + TILE_PIX * Y_STRIDE;        // 64 x CT_STRIDE
  float*  st_lds = (float*)(Ct_lds + C_CH * CT_STRIDE); // 128 floats

  const int tid      = threadIdx.x;
  const int pix_base = blockIdx.x * TILE_PIX;
  const int b        = pix_base / HW;
  const int hw0      = pix_base % HW;
  const int h        = hw0 / W_IMG;
  const int w0       = hw0 % W_IMG;   // multiple of 32: tile stays in one row

  if (tid < 128) st_lds[tid] = 0.0f;

#if USE_TDM
  // Wave 0 kicks off the async weight DMAs; they run concurrently with the
  // gather phase below. TENSORcnt is per-wave, so wave 0 waits, then the
  // workgroup barrier publishes the LDS panel to all waves.
  if (tid < 32) {
    tdm_load_1d((unsigned)(unsigned long long)(void*)Bt_lds, Wbt,
                (C_CH * BT_STRIDE * 2) / 8);            // 9216 x 8B
    tdm_load_1d((unsigned)(unsigned long long)(void*)Ct_lds, Wct,
                (C_CH * CT_STRIDE * 2) / 8);            // 1024 x 8B
  }
#else
  for (int i = tid; i < C_CH * BT_STRIDE; i += 256) Bt_lds[i] = Wbt[i];
  for (int i = tid; i < C_CH * CT_STRIDE; i += 256) Ct_lds[i] = Wct[i];
#endif

  const float* xb = x + (size_t)b * C_CH * HW;

  // Prefetch the x rows the (small-offset) bilinear taps will hit.
  {
    const int pc = tid & 63;
    int hh = h - 1 + (tid >> 6);
    hh = hh < 0 ? 0 : (hh >= H_IMG ? H_IMG - 1 : hh);
    __builtin_prefetch(xb + pc * HW + hh * W_IMG + w0, 0, 0);
  }

  // --- bilinear sampling: 32 pixels x 9 taps x 64 channels -> A panel -----
  const int pix  = tid & 31;    // lane == pixel -> spatially local gathers
  const int cblk = tid >> 5;    // 8 channel blocks of 8
  const int pw   = w0 + pix;
  const float* ob = offs + (size_t)b * 18 * HW + h * W_IMG + pw;
#pragma unroll
  for (int kk = 0; kk < KKT; ++kk) {
    const float oy = ob[(2 * kk) * HW];
    const float ox = ob[(2 * kk + 1) * HW];
    const float py = oy + (float)(h - 1 + kk / 3);
    const float px = ox + (float)(pw - 1 + kk % 3);
    const float fy = floorf(py), fx = floorf(px);
    const int iy0 = (int)fy, ix0 = (int)fx;
    const float wy = py - fy, wx = px - fx;
    const float w00 = (1.0f - wy) * (1.0f - wx);
    const float w01 = (1.0f - wy) * wx;
    const float w10 = wy * (1.0f - wx);
    const float w11 = wy * wx;
    const bool y0v = (iy0 >= 0) && (iy0 < H_IMG);
    const bool y1v = (iy0 + 1 >= 0) && (iy0 + 1 < H_IMG);
    const bool x0v = (ix0 >= 0) && (ix0 < W_IMG);
    const bool x1v = (ix0 + 1 >= 0) && (ix0 + 1 < W_IMG);
    const int base00 = iy0 * W_IMG + ix0;
#pragma unroll
    for (int i = 0; i < 8; ++i) {
      const int c = cblk * 8 + i;
      const float* xc = xb + c * HW;
      float v = 0.0f;
      if (y0v && x0v) v += w00 * xc[base00];
      if (y0v && x1v) v += w01 * xc[base00 + 1];
      if (y1v && x0v) v += w10 * xc[base00 + W_IMG];
      if (y1v && x1v) v += w11 * xc[base00 + W_IMG + 1];
      A_lds[pix * A_STRIDE + kk * 64 + c] = f2bf(v);
    }
  }

#if USE_TDM
  if (tid < 32) __builtin_amdgcn_s_wait_tensorcnt(0);
#endif
  __syncthreads();

  // --- GEMM1: y1[32x64] = A[32x576] * W_def^T, bf16 WMMA, f32 accum -------
  const int lane = tid & 31;
  const int wave = tid >> 5;
  const int mt = wave & 1;          // 2 M-tiles of 16 pixels
  const int nt = wave >> 1;         // 4 N-tiles of 16 channels
  const int l15 = lane & 15;
  const int hs  = lane >> 4;        // K-half select (this code's contiguous k = 16*hs + j)
  const int arow = mt * 16 + l15;   // A row (pixel)
  const int n    = nt * 16 + l15;   // output channel

  v8f acc = {0.f, 0.f, 0.f, 0.f, 0.f, 0.f, 0.f, 0.f};
  for (int ks = 0; ks < KTOT / 32; ++ks) {
    v16bf af, bf;
    const __bf16* ap = A_lds + arow * A_STRIDE + ks * 32 + hs * 16;
#pragma unroll
    for (int j = 0; j < 16; ++j) af[j] = ap[j];
    const __bf16* bp = Bt_lds + n * BT_STRIDE + ks * 32 + hs * 16;
#pragma unroll
    for (int j = 0; j < 16; ++j) bf[j] = bp[j];
    acc = __builtin_amdgcn_wmma_f32_16x16x32_bf16(
        false, af, false, bf, (short)0, acc, false, false);
  }
  {
    const float bd = b_def[n];
#pragma unroll
    for (int r = 0; r < 8; ++r) {
      const int m = mt * 16 + hs * 8 + r;
      Y_lds[m * Y_STRIDE + n] = f2bf(acc[r] + bd);
    }
  }
  __syncthreads();

  // --- GEMM2: y[32x64] = y1[32x64] * W_cat^T (K = 64, 2 WMMA steps) -------
  v8f acc2 = {0.f, 0.f, 0.f, 0.f, 0.f, 0.f, 0.f, 0.f};
  for (int ks = 0; ks < 2; ++ks) {
    v16bf af, bf;
    const __bf16* ap = Y_lds + arow * Y_STRIDE + ks * 32 + hs * 16;
#pragma unroll
    for (int j = 0; j < 16; ++j) af[j] = ap[j];
    const __bf16* bp = Ct_lds + n * CT_STRIDE + ks * 32 + hs * 16;
#pragma unroll
    for (int j = 0; j < 16; ++j) bf[j] = bp[j];
    acc2 = __builtin_amdgcn_wmma_f32_16x16x32_bf16(
        false, af, false, bf, (short)0, acc2, false, false);
  }

  // --- emit y (NHWC, f32) + per-channel BN partial sums -------------------
  {
    const float bc = b_cat[n];
    float ls = 0.0f, lq = 0.0f;
#pragma unroll
    for (int r = 0; r < 8; ++r) {
      const int m = mt * 16 + hs * 8 + r;
      const float v = acc2[r] + bc;
      y_ws[(size_t)(pix_base + m) * C_CH + n] = v;
      ls += v;
      lq += v * v;
    }
    atomicAdd(&st_lds[n], ls);        // ds_add_f32
    atomicAdd(&st_lds[64 + n], lq);
  }
  __syncthreads();
  if (tid < 128) atomicAdd(&stat_ws[tid], st_lds[tid]);
}

// ---------------------------------------------------------------------------
// Kernel 4: fold statistics into affine BN coefficients (64 threads).
// ---------------------------------------------------------------------------
__global__ void bn_finalize_kernel(const float* __restrict__ stat_ws,
                                   const float* __restrict__ gamma,
                                   const float* __restrict__ beta,
                                   float* __restrict__ coef_ws) {
  const int c = threadIdx.x;
  const float inv_n = 1.0f / (float)NPIX;
  const float mean = stat_ws[c] * inv_n;
  const float var  = stat_ws[64 + c] * inv_n - mean * mean;
  const float a    = gamma[c] * rsqrtf(var + 1e-5f);
  coef_ws[c]      = a;
  coef_ws[64 + c] = beta[c] - a * mean;
}

// ---------------------------------------------------------------------------
// Kernel 5: BN + ReLU + residual; NHWC y -> NCHW out via LDS transpose so
// both the y loads and the x/out accesses are fully coalesced.
// ---------------------------------------------------------------------------
__global__ __launch_bounds__(256) void apply_kernel(
    const float* __restrict__ y_ws, const float* __restrict__ coef_ws,
    const float* __restrict__ x,    float* __restrict__ out) {
  __shared__ float tl[64][65];
  __shared__ float ca[64], cb[64];
  const int tid = threadIdx.x;
  if (tid < 64) { ca[tid] = coef_ws[tid]; cb[tid] = coef_ws[64 + tid]; }

  const int pb = blockIdx.x * 64;          // 64-pixel tile, one image row seg
  for (int i = tid; i < 4096; i += 256) {
    const int p = i >> 6, ch = i & 63;
    tl[p][ch] = y_ws[(size_t)(pb + p) * C_CH + ch];
  }
  __syncthreads();

  const int b   = pb / HW;
  const int hwb = pb % HW;
  for (int i = tid; i < 4096; i += 256) {
    const int ch = i >> 6, p = i & 63;
    float v = ca[ch] * tl[p][ch] + cb[ch];
    v = v > 0.0f ? v : 0.0f;
    const size_t idx = ((size_t)b * C_CH + ch) * HW + hwb + p;
    out[idx] = v + x[idx];
  }
}

// ---------------------------------------------------------------------------
// Host-side launcher.
// ---------------------------------------------------------------------------
extern "C" void kernel_launch(void* const* d_in, const int* in_sizes, int n_in,
                              void* d_out, int out_size, void* d_ws, size_t ws_size,
                              hipStream_t stream) {
  const float* x     = (const float*)d_in[0];
  const float* W_off = (const float*)d_in[1];
  const float* b_off = (const float*)d_in[2];
  const float* W_def = (const float*)d_in[3];
  const float* b_def = (const float*)d_in[4];
  const float* W_cat = (const float*)d_in[5];
  const float* b_cat = (const float*)d_in[6];
  const float* gamma = (const float*)d_in[7];
  const float* beta  = (const float*)d_in[8];
  float* out = (float*)d_out;

  // workspace layout
  float* ws      = (float*)d_ws;
  float* offs    = ws;                       // 4*18*36864    = 2,654,208 f32
  float* y_ws    = offs + 2654208;           // 147456*64     = 9,437,184 f32
  float* stat_ws = y_ws + 9437184;           // 128 f32
  float* coef_ws = stat_ws + 128;            // 128 f32
  __bf16* Wbt    = (__bf16*)(coef_ws + 128); // 64*576 bf16 (16B-aligned)
  __bf16* Wct    = Wbt + C_CH * KTOT;        // 64*64  bf16

  hipLaunchKernelGGL(zero_stats_kernel, dim3(1), dim3(128), 0, stream, stat_ws);

  hipLaunchKernelGGL(prep_weights_kernel, dim3(160), dim3(256), 0, stream,
                     W_def, W_cat, Wbt, Wct);

  hipLaunchKernelGGL(offset_conv_kernel, dim3(B_N * 18 * (HW / 256)), dim3(256),
                     0, stream, x, W_off, b_off, offs);

  // dynamic LDS: A(32x584) + Bt(64x576) + Y(32x72) + Ct(64x64) bf16 + 128 f32
  const size_t smem = (size_t)(TILE_PIX * A_STRIDE + C_CH * BT_STRIDE +
                               TILE_PIX * Y_STRIDE + C_CH * CT_STRIDE) * 2 +
                      128 * 4;
  hipLaunchKernelGGL(deform_gemm_kernel, dim3(NPIX / TILE_PIX), dim3(256),
                     smem, stream, x, offs, Wbt, b_def, Wct, b_cat,
                     y_ws, stat_ws);

  hipLaunchKernelGGL(bn_finalize_kernel, dim3(1), dim3(64), 0, stream,
                     stat_ws, gamma, beta, coef_ws);

  hipLaunchKernelGGL(apply_kernel, dim3(NPIX / 64), dim3(256), 0, stream,
                     y_ws, coef_ws, x, out);
}
